// TSGCNet_46935402611410
// MI455X (gfx1250) — matrix operand order repaired and checked by reading, not compiled, not run on hardware
//
#include <hip/hip_runtime.h>
#include <hip/hip_bf16.h>
#include <cstdint>
#include <cstddef>

typedef __attribute__((ext_vector_type(16))) _Float16 v16h;
typedef __attribute__((ext_vector_type(8)))  float    v8f;

#define MODE_DENSE   0
#define MODE_CONCAT2 1
#define MODE_CONCAT3 2
#define MODE_MUL2    3
#define MODE_EDGE    4
#define MODE_DELTA   5

#define ACT_NONE  0
#define ACT_RELU  1
#define ACT_LRELU 2
#define ACT_SIG   3
#define ACT_TANH  4

#define NPTS 4096
#define NBATCH 2

struct XSrc {
  const float* x1; const float* x2; const float* x3;
  const int*   nidx;
  long long bs1, cs1, bs2, cs2, bs3, cs3;
  int C1, C2;
  int mode, Kn, istride, npts;
};

struct Epi {
  const float* bias;
  const float* bnb; const float* bng; const float* bnm; const float* bnv;
  int act;
};

// ch/col must already be clamped in-range; always executes its loads.
__device__ __forceinline__ float xsrc_load(const XSrc& s, int b, int ch, int col) {
  switch (s.mode) {
    case MODE_DENSE:
      return s.x1[(long long)b * s.bs1 + (long long)ch * s.cs1 + col];
    case MODE_CONCAT2:
      if (ch < s.C1)
        return s.x1[(long long)b * s.bs1 + (long long)ch * s.cs1 + col];
      return s.x2[(long long)b * s.bs2 + (long long)(ch - s.C1) * s.cs2 + col];
    case MODE_CONCAT3:
      if (ch < s.C1)
        return s.x1[(long long)b * s.bs1 + (long long)ch * s.cs1 + col];
      if (ch < s.C1 + s.C2)
        return s.x2[(long long)b * s.bs2 + (long long)(ch - s.C1) * s.cs2 + col];
      return s.x3[(long long)b * s.bs3 + (long long)(ch - s.C1 - s.C2) * s.cs3 + col];
    case MODE_MUL2:
      return s.x1[(long long)b * s.bs1 + (long long)ch * s.cs1 + col] *
             s.x2[(long long)b * s.bs2 + (long long)ch * s.cs2 + col];
    case MODE_EDGE: {   // concat([nb - c, c])
      int n = col / s.Kn, k = col % s.Kn;
      int nb = s.nidx[((long long)b * s.npts + n) * s.istride + k];
      if (ch < s.C1)
        return s.x1[(long long)b * s.bs1 + (long long)ch * s.cs1 + nb] -
               s.x1[(long long)b * s.bs1 + (long long)ch * s.cs1 + n];
      return s.x1[(long long)b * s.bs1 + (long long)(ch - s.C1) * s.cs1 + n];
    }
    case MODE_DELTA: {  // concat([c - nb, nb])
      int n = col / s.Kn, k = col % s.Kn;
      int nb = s.nidx[((long long)b * s.npts + n) * s.istride + k];
      if (ch < s.C1)
        return s.x1[(long long)b * s.bs1 + (long long)ch * s.cs1 + n] -
               s.x1[(long long)b * s.bs1 + (long long)ch * s.cs1 + nb];
      return s.x1[(long long)b * s.bs1 + (long long)(ch - s.C1) * s.cs1 + nb];
    }
  }
  return 0.f;
}

__device__ __forceinline__ float apply_act(float v, int act) {
  switch (act) {
    case ACT_RELU:  return fmaxf(v, 0.f);
    case ACT_LRELU: return v > 0.f ? v : 0.2f * v;
    case ACT_SIG:   return 1.f / (1.f + expf(-v));
    case ACT_TANH:  return tanhf(v);
  }
  return v;
}

// ---------------------------------------------------------------------------
// Fused GEMM:  out[b,o,col] = act(BN(W[o,:] . X[b,:,col] + bias[o]))
// W: (O, C) fp32 row-major.  X supplied via XSrc functor (gathers fused in).
// 128 threads = 4 waves, 64x64 tile, each wave 32x32 via 2x2 WMMA f16 frags.
// LDS tiles are stored in *fragment order* so each lane's v16h fragment is a
// single contiguous 32B chunk (2x ds_load_b128, no v_mov_b16 packing):
//   As: per output-row, 32 k-halves contiguous [0..15 | 16..31]
//   Bs: column-major, 32 k-halves contiguous per column (stride 48 halves)
// ---------------------------------------------------------------------------
__global__ __launch_bounds__(128)
void wmma_gemm_kernel(const float* __restrict__ W, XSrc xs, Epi ep,
                      float* __restrict__ out, int O, int C, int cols)
{
  __shared__ __align__(32) _Float16 As[64 * 32];   // [row][k]
  __shared__ __align__(32) _Float16 Bs[64 * 48];   // [col][k] (+pad to 48)

  const int b  = blockIdx.z;
  const int o0 = blockIdx.y * 64;
  const int c0 = blockIdx.x * 64;
  const int t    = threadIdx.x;
  const int lane = t & 31;
  const int wv   = t >> 5;
  const int wr   = wv >> 1;        // wave row (0..1)
  const int wc   = wv & 1;         // wave col (0..1)
  const int half = lane >> 4;
  const int mm   = lane & 15;

  v8f acc00, acc01, acc10, acc11;
#pragma unroll
  for (int r = 0; r < 8; ++r) { acc00[r] = 0.f; acc01[r] = 0.f; acc10[r] = 0.f; acc11[r] = 0.f; }

  for (int k0 = 0; k0 < C; k0 += 32) {
    // ---- stage W tile (64 x 32): batched unconditional loads, then convert ----
    float wvv[16];
#pragma unroll
    for (int j = 0; j < 16; ++j) {
      int i = t + j * 128;
      int r = i >> 5, c = i & 31;
      int oo = o0 + r, kk = k0 + c;
      bool ok = (oo < O) & (kk < C);
      int oc = oo < O ? oo : O - 1;
      int kc = kk < C ? kk : C - 1;
      float v = W[(long long)oc * C + kc];
      wvv[j] = ok ? v : 0.f;
    }
#pragma unroll
    for (int j = 0; j < 16; ++j) {
      int i = t + j * 128;
      int r = i >> 5, c = i & 31;
      int pos = c;
      As[r * 32 + pos] = (_Float16)wvv[j];
    }
    // ---- stage X tile (32 x 64): gather fused, batched loads ----
    float xvv[16];
#pragma unroll
    for (int j = 0; j < 16; ++j) {
      int i = t + j * 128;
      int r = i >> 6, c = i & 63;        // r = k-local, c = col-local
      int kk = k0 + r, col = c0 + c;
      bool ok = (kk < C) & (col < cols);
      int kc = kk < C ? kk : C - 1;
      int cc = col < cols ? col : cols - 1;
      float v = xsrc_load(xs, b, kc, cc);
      xvv[j] = ok ? v : 0.f;
    }
#pragma unroll
    for (int j = 0; j < 16; ++j) {
      int i = t + j * 128;
      int r = i >> 6, c = i & 63;
      Bs[c * 48 + r] = (_Float16)xvv[j];
    }
    __syncthreads();

    // ---- fragments: contiguous 32B LDS chunks ----
    v16h a0  = *(const v16h*)&As[(wr * 32 + mm) * 32 + 16 * half];
    v16h a1  = *(const v16h*)&As[(wr * 32 + 16 + mm) * 32 + 16 * half];
    v16h bv0 = *(const v16h*)&Bs[(wc * 32 + mm) * 48 + 16 * half];
    v16h bv1 = *(const v16h*)&Bs[(wc * 32 + 16 + mm) * 48 + 16 * half];

    acc00 = __builtin_amdgcn_wmma_f32_16x16x32_f16(false, a0, false, bv0, (short)0, acc00, false, false);
    acc01 = __builtin_amdgcn_wmma_f32_16x16x32_f16(false, a0, false, bv1, (short)0, acc01, false, false);
    acc10 = __builtin_amdgcn_wmma_f32_16x16x32_f16(false, a1, false, bv0, (short)0, acc10, false, false);
    acc11 = __builtin_amdgcn_wmma_f32_16x16x32_f16(false, a1, false, bv1, (short)0, acc11, false, false);
    __syncthreads();
  }

  // epilogue + store.  C/D layout: VGPR r, lane L -> M = r + 8*(L/16), Ncol = L%16
#pragma unroll
  for (int i = 0; i < 2; ++i) {
#pragma unroll
    for (int j = 0; j < 2; ++j) {
      const v8f& acc = (i == 0) ? (j == 0 ? acc00 : acc01) : (j == 0 ? acc10 : acc11);
#pragma unroll
      for (int r = 0; r < 8; ++r) {
        int oo  = o0 + wr * 32 + i * 16 + r + 8 * half;
        int col = c0 + wc * 32 + j * 16 + mm;
        if (oo < O && col < cols) {
          float v = acc[r];
          if (ep.bias) v += ep.bias[oo];
          if (ep.bng) {
            float sc = ep.bng[oo] * rsqrtf(ep.bnv[oo] + 1e-5f);
            v = (v - ep.bnm[oo]) * sc + ep.bnb[oo];
          }
          out[((long long)b * O + oo) * cols + col] = apply_act(v, ep.act);
        }
      }
    }
  }
}

// ---------------------------------------------------------------------------
// KNN: one thread per query point; insertion-sorted top-33 (incl. self) in LDS.
// ---------------------------------------------------------------------------
__global__ __launch_bounds__(128)
void knn_kernel(const float* __restrict__ x, int* __restrict__ idx32)
{
  __shared__ float sd[128 * 33];
  __shared__ int   si[128 * 33];
  int q = blockIdx.x * 128 + threadIdx.x;
  if (q >= NBATCH * NPTS) return;
  int b = q / NPTS, n = q % NPTS;
  float* dl = &sd[threadIdx.x * 33];
  int*   il = &si[threadIdx.x * 33];
  for (int k = 0; k < 33; ++k) { dl[k] = 3.4e38f; il[k] = 0; }
  const float* px = x + (long long)b * 6 * NPTS;
  float qx = px[n], qy = px[NPTS + n], qz = px[2 * NPTS + n];
  for (int m = 0; m < NPTS; ++m) {
    float dx = px[m] - qx, dy = px[NPTS + m] - qy, dz = px[2 * NPTS + m] - qz;
    float d = dx * dx + dy * dy + dz * dz;
    if (d < dl[32]) {
      int p = 32;
      while (p > 0 && dl[p - 1] > d) { dl[p] = dl[p - 1]; il[p] = il[p - 1]; --p; }
      dl[p] = d; il[p] = m;
    }
  }
  for (int k = 0; k < 32; ++k)
    idx32[((long long)b * NPTS + n) * 32 + k] = il[k + 1];
}

__global__ void maxpool_kernel(const float* __restrict__ in, float* __restrict__ out, int rows, int n)
{
  int i = blockIdx.x * blockDim.x + threadIdx.x;
  if (i >= rows) return;
  float m = -3.4e38f;
  for (int j = 0; j < n; ++j) m = fmaxf(m, in[(long long)i * n + j]);
  out[i] = m;
}

__global__ void fc_kernel(const float* __restrict__ W, const float* __restrict__ bias,
                          const float* __restrict__ in, float* __restrict__ out,
                          int B, int O, int C, int act, int addIdentity3)
{
  int i = blockIdx.x * blockDim.x + threadIdx.x;
  if (i >= B * O) return;
  int b = i / O, o = i % O;
  float s = bias ? bias[o] : 0.f;
  for (int c = 0; c < C; ++c) s += W[(long long)o * C + c] * in[(long long)b * C + c];
  s = apply_act(s, act);
  if (addIdentity3 && (o / 3) == (o % 3)) s += 1.f;
  out[i] = s;
}

// coor_new[b,i,n] = sum_j x[b,j,n] * T[b, j*3+i]
__global__ void apply_T3_kernel(const float* __restrict__ x, long long bs, long long cs,
                                const float* __restrict__ T, float* __restrict__ out)
{
  int i = blockIdx.x * blockDim.x + threadIdx.x;
  if (i >= NBATCH * NPTS) return;
  int b = i / NPTS, n = i % NPTS;
  float p0 = x[(long long)b * bs + 0 * cs + n];
  float p1 = x[(long long)b * bs + 1 * cs + n];
  float p2 = x[(long long)b * bs + 2 * cs + n];
  for (int c = 0; c < 3; ++c)
    out[((long long)b * 3 + c) * NPTS + n] =
        p0 * T[b * 9 + 0 * 3 + c] + p1 * T[b * 9 + 1 * 3 + c] + p2 * T[b * 9 + 2 * 3 + c];
}

// out[b,o,n] = sum_k softmax_k(e[b,o,n*K+k]) * f[b,o,n*K+k]
__global__ void ga_combine_kernel(const float* __restrict__ e, const float* __restrict__ f,
                                  float* __restrict__ out, int O, int K)
{
  int i = blockIdx.x * blockDim.x + threadIdx.x;
  if (i >= NBATCH * O * NPTS) return;
  int n = i % NPTS;
  long long row = i / NPTS;                   // b*O + o
  const float* ep = e + row * (long long)NPTS * K + (long long)n * K;
  const float* fp = f + row * (long long)NPTS * K + (long long)n * K;
  float mx = -3.4e38f;
  for (int k = 0; k < K; ++k) mx = fmaxf(mx, ep[k]);
  float s = 0.f, acc = 0.f;
  for (int k = 0; k < K; ++k) { float w = expf(ep[k] - mx); s += w; acc += w * fp[k]; }
  out[row * NPTS + n] = acc / s;
}

// coef[b,n,k] = softmax_k( theta(x_n) . theta(x_nb) )
__global__ void nl_coef_kernel(const float* __restrict__ t, const int* __restrict__ idx32,
                               float* __restrict__ coef, int Ci, int K)
{
  int i = blockIdx.x * blockDim.x + threadIdx.x;
  if (i >= NBATCH * NPTS) return;
  int b = i / NPTS, n = i % NPTS;
  float lg[32];
  float mx = -3.4e38f;
  for (int k = 0; k < K; ++k) {
    int nb = idx32[((long long)b * NPTS + n) * 32 + k];
    float s = 0.f;
    for (int c = 0; c < Ci; ++c)
      s += t[((long long)b * Ci + c) * NPTS + n] * t[((long long)b * Ci + c) * NPTS + nb];
    lg[k] = s;
    mx = fmaxf(mx, s);
  }
  float sum = 0.f;
  for (int k = 0; k < K; ++k) { lg[k] = expf(lg[k] - mx); sum += lg[k]; }
  float inv = 1.f / sum;
  for (int k = 0; k < K; ++k) coef[((long long)b * NPTS + n) * K + k] = lg[k] * inv;
}

// out[b,o,n] = sum_k coef[b,n,k] * g[b,o,n*K+k]
__global__ void nl_combine_kernel(const float* __restrict__ coef, const float* __restrict__ g,
                                  float* __restrict__ out, int O, int K)
{
  int i = blockIdx.x * blockDim.x + threadIdx.x;
  if (i >= NBATCH * O * NPTS) return;
  int n = i % NPTS;
  long long row = i / NPTS;                   // b*O + o
  int b = (int)(row / O);
  const float* gp = g + row * (long long)NPTS * K + (long long)n * K;
  const float* cp = coef + ((long long)b * NPTS + n) * K;
  float s = 0.f;
  for (int k = 0; k < K; ++k) s += cp[k] * gp[k];
  out[row * NPTS + n] = s;
}

__global__ void softmax_rows_kernel(float* __restrict__ a, int rows, int n)
{
  int r = blockIdx.x * blockDim.x + threadIdx.x;
  if (r >= rows) return;
  float* p = a + (long long)r * n;
  float mx = -3.4e38f;
  for (int i = 0; i < n; ++i) mx = fmaxf(mx, p[i]);
  float s = 0.f;
  for (int i = 0; i < n; ++i) { float e = expf(p[i] - mx); p[i] = e; s += e; }
  float inv = 1.f / s;
  for (int i = 0; i < n; ++i) p[i] *= inv;
}

// M[b,i,c] = sum_n A[b,i,n] * H[b,c,n]
__global__ void pool_M_kernel(const float* __restrict__ A, const float* __restrict__ H,
                              float* __restrict__ M)
{
  int i = blockIdx.x * blockDim.x + threadIdx.x;
  if (i >= NBATCH * 14 * 1024) return;
  int b = i / (14 * 1024);
  int rem = i % (14 * 1024);
  int r = rem / 1024, c = rem % 1024;
  float s = 0.f;
  for (int n = 0; n < NPTS; ++n)
    s += A[((long long)b * 14 + r) * NPTS + n] * H[((long long)b * 1024 + c) * NPTS + n];
  M[i] = s;
}

__global__ void final_out_kernel(const float* __restrict__ M, const float* __restrict__ pqw,
                                 float* __restrict__ out)
{
  int i = blockIdx.x * blockDim.x + threadIdx.x;
  if (i >= NBATCH * 14) return;
  int b = i / 14, r = i % 14;
  float s = 0.f;
  for (int c = 0; c < 1024; ++c)
    s += M[((long long)b * 14 + r) * 1024 + c] * pqw[(long long)r * 1024 + c];
  out[i] = s;
}

// ---------------------------------------------------------------------------
// Host side
// ---------------------------------------------------------------------------
static XSrc XDenseStr(const float* x, long long bs, long long cs) {
  XSrc s{}; s.mode = MODE_DENSE; s.x1 = x; s.bs1 = bs; s.cs1 = cs; return s;
}
static XSrc XDense(const float* x, int C, long long cols) {
  return XDenseStr(x, (long long)C * cols, cols);
}
static XSrc XCat2(const float* a, int Ca, const float* b, int Cb, long long cols) {
  XSrc s{}; s.mode = MODE_CONCAT2;
  s.x1 = a; s.bs1 = (long long)Ca * cols; s.cs1 = cols;
  s.x2 = b; s.bs2 = (long long)Cb * cols; s.cs2 = cols;
  s.C1 = Ca; s.C2 = Cb; return s;
}
static XSrc XCat3(const float* a, int Ca, const float* b, int Cb, const float* c, int Cc, long long cols) {
  XSrc s{}; s.mode = MODE_CONCAT3;
  s.x1 = a; s.bs1 = (long long)Ca * cols; s.cs1 = cols;
  s.x2 = b; s.bs2 = (long long)Cb * cols; s.cs2 = cols;
  s.x3 = c; s.bs3 = (long long)Cc * cols; s.cs3 = cols;
  s.C1 = Ca; s.C2 = Cb; return s;
}
static XSrc XMul2(const float* a, const float* b, int C, long long cols) {
  XSrc s{}; s.mode = MODE_MUL2;
  s.x1 = a; s.bs1 = (long long)C * cols; s.cs1 = cols;
  s.x2 = b; s.bs2 = (long long)C * cols; s.cs2 = cols;
  return s;
}
static XSrc XGather(int mode, const float* x, int C1, long long bs, long long cs,
                    const int* idx, int K) {
  XSrc s{}; s.mode = mode; s.x1 = x; s.bs1 = bs; s.cs1 = cs;
  s.C1 = C1; s.nidx = idx; s.Kn = K; s.istride = 32; s.npts = NPTS; return s;
}

static void gemm(hipStream_t st, const float* W, const XSrc& xs, const Epi& ep,
                 float* out, int O, int C, long long cols)
{
  dim3 grid((unsigned)((cols + 63) / 64), (unsigned)((O + 63) / 64), NBATCH);
  wmma_gemm_kernel<<<grid, 128, 0, st>>>(W, xs, ep, out, O, C, (int)cols);
}

// param-leaf indices: jax pytree flatten (dict keys sorted) of setup_inputs().
enum {
  pAMC = 0, pAMN = 18,
  pAU_B = 36, pAU_W, pAV_B, pAV_W, pAW_B, pAW_W,
  pC1C = 42, pC1N = 47, pC2C = 52, pC2N = 57, pC3C = 62, pC3N = 67, pC5C = 72, pC5N = 77,
  pFTMC = 82, pFTMN = 106,
  pGA1 = 130, pGA2 = 135, pGA3 = 140,
  pNL1 = 145, pNL2 = 155, pNL3 = 165,
  pPQ_W = 175, pPQS = 176, pX = 181
};
// within-block offsets (keys sorted, bn expanded as b,g,m,v)
enum { aB1 = 0, aB2, aB3, aBN1 = 3, aBN2 = 7, aBN3 = 11, aW1 = 15, aW2, aW3 };              // attm
enum { sB1 = 0, sB2, sB3, sBN1 = 3, sBN2 = 7, sBN3 = 11,
       sFB1 = 15, sFB2, sFB3, sFC1 = 18, sFC2, sFC3, sW1 = 21, sW2, sW3 };                   // stn
enum { cBN = 0, cW = 4 };                                                                    // conv_bn
enum { nBG = 0, nBN = 1, nBT = 5, nBW = 6, nWG = 7, nWT = 8, nWW = 9 };                      // nl

extern "C" void kernel_launch(void* const* d_in, const int* in_sizes, int n_in,
                              void* d_out, int out_size, void* d_ws, size_t ws_size,
                              hipStream_t stream)
{
  const long long N = NPTS;
  auto F = [&](int i) -> const float* { return (const float*)d_in[i]; };
  const float* xin = F(pX);            // (2, 6, 4096)

  // ---- workspace carve-out ----
  char* ws = (char*)d_ws;
  size_t off = 0;
  auto alloc = [&](size_t bytes) -> char* {
    char* p = ws + off; off = (off + bytes + 255) & ~(size_t)255; return p;
  };
  int*   idx32  = (int*)  alloc(2ull * N * 32 * 4);
  float* coorT  = (float*)alloc(2ull * 3 * N * 4);
  float* norT   = (float*)alloc(2ull * 3 * N * 4);
  float* sh1    = (float*)alloc(2ull * 64 * N * 4);
  float* sh2    = (float*)alloc(2ull * 128 * N * 4);
  float* sh3    = (float*)alloc(2ull * 1024 * N * 4);
  float* pool   = (float*)alloc(2ull * 1024 * 4);
  float* f1     = (float*)alloc(2ull * 512 * 4);
  float* f2     = (float*)alloc(2ull * 256 * 4);
  float* Tc     = (float*)alloc(256);
  float* Tn     = (float*)alloc(256);
  float* coor1  = (float*)alloc(2ull * 64 * N * 4);
  float* nor1   = (float*)alloc(2ull * 64 * N * 4);
  float* coor2  = (float*)alloc(2ull * 64 * N * 4);
  float* nor2   = (float*)alloc(2ull * 64 * N * 4);
  float* coor3  = (float*)alloc(2ull * 128 * N * 4);
  float* nor3   = (float*)alloc(2ull * 128 * N * 4);
  float* coor3a = (float*)alloc(2ull * 256 * N * 4);
  float* nor3a  = (float*)alloc(2ull * 256 * N * 4);
  float* attm1  = (float*)alloc(2ull * 256 * N * 4);
  float* attm2  = (float*)alloc(2ull * 128 * N * 4);
  float* cc     = (float*)alloc(2ull * 512 * N * 4);
  float* nc     = (float*)alloc(2ull * 512 * N * 4);
  float* hbuf   = (float*)alloc(2ull * 1024 * N * 4);
  float* tbuf   = (float*)alloc(2ull * 64 * N * 4);
  float* nltmp  = (float*)alloc(2ull * 128 * N * 4);
  float* coef   = (float*)alloc(2ull * N * 32 * 4);
  float* a14    = (float*)alloc(2ull * 14 * N * 4);
  float* mbuf   = (float*)alloc(2ull * 14 * 1024 * 4);
  float* bigA   = (float*)alloc(2ull * 128 * N * 32 * 4);   // 134 MB
  float* bigB   = (float*)alloc(2ull * 128 * N * 32 * 4);   // 134 MB
  float* AVb = bigA;   // reused after stage 3
  float* AUb = bigB;

  auto EP = [&](const float* bias, int bnBase, int act) -> Epi {
    Epi e{}; e.bias = bias;
    if (bnBase >= 0) { e.bnb = F(bnBase); e.bng = F(bnBase + 1); e.bnm = F(bnBase + 2); e.bnv = F(bnBase + 3); }
    e.act = act; return e;
  };

  // ---- 1. KNN on original coordinates ----
  knn_kernel<<<(NBATCH * NPTS + 127) / 128, 128, 0, stream>>>(xin, idx32);

  // ---- 2. STNs ----
  auto runStn = [&](int P, const float* xbase, float* Tout) {
    gemm(stream, F(P + sW1), XDenseStr(xbase, 6 * N, N), EP(F(P + sB1), P + sBN1, ACT_RELU), sh1, 64, 3, N);
    gemm(stream, F(P + sW2), XDense(sh1, 64, N),  EP(F(P + sB2), P + sBN2, ACT_RELU), sh2, 128, 64, N);
    gemm(stream, F(P + sW3), XDense(sh2, 128, N), EP(F(P + sB3), P + sBN3, ACT_RELU), sh3, 1024, 128, N);
    maxpool_kernel<<<(2048 + 255) / 256, 256, 0, stream>>>(sh3, pool, 2 * 1024, NPTS);
    fc_kernel<<<4, 256, 0, stream>>>(F(P + sFC1), F(P + sFB1), pool, f1, 2, 512, 1024, ACT_RELU, 0);
    fc_kernel<<<2, 256, 0, stream>>>(F(P + sFC2), F(P + sFB2), f1, f2, 2, 256, 512, ACT_RELU, 0);
    fc_kernel<<<1, 32, 0, stream>>>(F(P + sFC3), F(P + sFB3), f2, Tout, 2, 9, 256, ACT_NONE, 1);
  };
  runStn(pFTMC, xin, Tc);
  apply_T3_kernel<<<(NBATCH * NPTS + 255) / 256, 256, 0, stream>>>(xin, 6 * N, N, Tc, coorT);
  runStn(pFTMN, xin + 3 * N, Tn);
  apply_T3_kernel<<<(NBATCH * NPTS + 255) / 256, 256, 0, stream>>>(xin + 3 * N, 6 * N, N, Tn, norT);

  // ---- helpers for graph stages ----
  auto gaStage = [&](int pConv, int pGA, const float* xpt, int Cin, int Oc, int K,
                     float* outb) {
    long long cols = N * K;
    gemm(stream, F(pConv + cW), XGather(MODE_EDGE, xpt, Cin, (long long)Cin * N, N, idx32, K),
         EP(nullptr, pConv + cBN, ACT_LRELU), bigA, Oc, 2 * Cin, cols);
    gemm(stream, F(pGA + cBN + 4), XGather(MODE_DELTA, xpt, Cin, (long long)Cin * N, N, idx32, K),
         EP(nullptr, pGA + cBN, ACT_LRELU), bigB, Oc, 2 * Cin, cols);
    ga_combine_kernel<<<(unsigned)((2ull * Oc * N + 255) / 256), 256, 0, stream>>>(bigB, bigA, outb, Oc, K);
  };
  auto nlStage = [&](int pConv, int pNL, const float* xpt, int Ci, int Co, int K,
                     const float* feat_src, int CinF, float* outb) {
    long long cols = N * K;
    gemm(stream, F(pConv + cW), XGather(MODE_EDGE, feat_src, CinF, (long long)CinF * N, N, idx32, K),
         EP(nullptr, pConv + cBN, ACT_LRELU), bigA, Co, 2 * CinF, cols);
    gemm(stream, F(pNL + nWT), XDense(xpt, Ci, N), EP(F(pNL + nBT), -1, ACT_NONE), tbuf, Ci, Ci, N);
    gemm(stream, F(pNL + nWG), XDense(bigA, Co, cols), EP(F(pNL + nBG), -1, ACT_NONE), bigB, Co, Co, cols);
    nl_coef_kernel<<<(NBATCH * NPTS + 255) / 256, 256, 0, stream>>>(tbuf, idx32, coef, Ci, K);
    nl_combine_kernel<<<(unsigned)((2ull * Co * N + 255) / 256), 256, 0, stream>>>(coef, bigB, nltmp, Co, K);
    gemm(stream, F(pNL + nWW), XDense(nltmp, Co, N), EP(F(pNL + nBW), pNL + nBN, ACT_NONE), outb, Co, Co, N);
  };

  // ---- 3. stage 1 (K=16) ----
  gaStage(pC1C, pGA1, coorT, 3, 64, 16, coor1);
  nlStage(pC1N, pNL1, norT, 3, 64, 16, norT, 3, nor1);
  // ---- 4. stage 2 (K=16) ----
  gaStage(pC2C, pGA2, coor1, 64, 64, 16, coor2);
  nlStage(pC2N, pNL2, nor1, 64, 64, 16, nor1, 64, nor2);
  // ---- 5. stage 3 (K=32) ----
  gaStage(pC3C, pGA3, coor2, 64, 128, 32, coor3);
  nlStage(pC3N, pNL3, nor2, 64, 128, 32, nor2, 64, nor3);

  // ---- 6. cross-attention modules ----
  auto runAtt = [&](int P, const float* x1, const float* x2, float* outb) {
    gemm(stream, F(P + aW1), XCat2(x1, 128, x2, 128, N), EP(F(P + aB1), P + aBN1, ACT_RELU), attm1, 256, 256, N);
    gemm(stream, F(P + aW2), XDense(attm1, 256, N),      EP(F(P + aB2), P + aBN2, ACT_SIG),  attm2, 128, 256, N);
    gemm(stream, F(P + aW3), XMul2(attm2, x1, 128, N),   EP(F(P + aB3), P + aBN3, ACT_RELU), outb, 256, 128, N);
  };
  runAtt(pAMC, coor3, nor3, coor3a);
  runAtt(pAMN, nor3, coor3, nor3a);

  // ---- 7. trunk ----
  gemm(stream, F(pC5C + cW), XCat3(coor1, 64, coor2, 64, coor3a, 256, N),
       EP(nullptr, pC5C + cBN, ACT_LRELU), cc, 512, 384, N);
  gemm(stream, F(pC5N + cW), XCat3(nor1, 64, nor2, 64, nor3a, 256, N),
       EP(nullptr, pC5N + cBN, ACT_LRELU), nc, 512, 384, N);
  gemm(stream, F(pPQS + cW), XCat2(cc, 512, nc, 512, N),
       EP(nullptr, pPQS + cBN, ACT_LRELU), hbuf, 1024, 1024, N);

  // ---- 8. attention pooling head ----
  gemm(stream, F(pAV_W), XDense(hbuf, 1024, N), EP(F(pAV_B), -1, ACT_TANH), AVb, 512, 1024, N);
  gemm(stream, F(pAU_W), XDense(hbuf, 1024, N), EP(F(pAU_B), -1, ACT_SIG),  AUb, 512, 1024, N);
  gemm(stream, F(pAW_W), XMul2(AVb, AUb, 512, N), EP(F(pAW_B), -1, ACT_NONE), a14, 14, 512, N);
  softmax_rows_kernel<<<1, 32, 0, stream>>>(a14, NBATCH * 14, NPTS);
  pool_M_kernel<<<(NBATCH * 14 * 1024 + 255) / 256, 256, 0, stream>>>(a14, hbuf, mbuf);
  final_out_kernel<<<1, 32, 0, stream>>>(mbuf, F(pPQ_W), (float*)d_out);

  (void)in_sizes; (void)n_in; (void)out_size; (void)ws_size;
}
